// GraphConvolutionalBlock_21500606284453
// MI455X (gfx1250) — matrix operand order repaired and checked by reading, not compiled, not run on hardware
//
#include <hip/hip_runtime.h>
#include <hip/hip_bf16.h>

#define D_HID 128

typedef __attribute__((ext_vector_type(16))) __bf16 v16bf;
typedef __attribute__((ext_vector_type(8)))  float  v8f;

union ABFrag { uint4 u[2]; v16bf v; };
union BF4    { __bf16 h[4]; uint2 u; };

// ---------------------------------------------------------------------------
// f32 -> bf16 bulk convert (weights + features)
// ---------------------------------------------------------------------------
__global__ __launch_bounds__(256) void k_cvt_bf16(const float* __restrict__ src,
                                                  __bf16* __restrict__ dst,
                                                  int count4) {
  int t = blockIdx.x * blockDim.x + threadIdx.x;
  if (t >= count4) return;
  float4 f = reinterpret_cast<const float4*>(src)[t];
  BF4 o;
  o.h[0] = (__bf16)f.x; o.h[1] = (__bf16)f.y;
  o.h[2] = (__bf16)f.z; o.h[3] = (__bf16)f.w;
  reinterpret_cast<uint2*>(dst)[t] = o.u;
}

// ---------------------------------------------------------------------------
// Fused dual GEMM: y = x@W0^T + b0 ; n = x@W1^T + b1   (x bf16, out f32)
// Block: 256 threads (8 waves). Block covers 16 rows; wave w covers cols
// [16w, 16w+16). K = 128 -> 4 WMMA steps per output matrix (8 WMMA total).
// ---------------------------------------------------------------------------
__global__ __launch_bounds__(256) void k_gemm_dual(
    const __bf16* __restrict__ xb,
    const __bf16* __restrict__ w0b, const __bf16* __restrict__ w1b,
    const float* __restrict__ b0,  const float* __restrict__ b1,
    float* __restrict__ ybuf, float* __restrict__ nbuf, int nrows)
{
  // 16 rows x 128 K of bf16, padded to 136 per row to avoid LDS bank conflicts
  __shared__ __align__(16) __bf16 atile[16 * 136];

  const int tid  = threadIdx.x;
  const int wave = tid >> 5;        // 0..7 -> column tile
  const int lane = tid & 31;
  const int row0 = blockIdx.x << 4;

  // Stage A tile: 2048 bf16 = 256 threads x 8 bf16 (one b128 each)
  {
    const int r   = tid >> 4;       // 0..15
    const int seg = tid & 15;       // 0..15 (8 bf16 each)
    int rr = row0 + r; if (rr >= nrows) rr = nrows - 1;   // clamp tail reads
    const uint4 v = *reinterpret_cast<const uint4*>(
        xb + (size_t)rr * D_HID + seg * 8);
    *reinterpret_cast<uint4*>(&atile[r * 136 + seg * 8]) = v;
  }
  __syncthreads();

  const int ncol = (wave << 4) | (lane & 15);  // global output column
  const int m    = lane & 15;                  // A-matrix row for this lane
  const int hi   = lane >> 4;                  // lane half
  const int khA  = hi * 16;                    // A frag: lanes 16-31 hold K 16-31
  const int khB  = hi * 16;                    // B frag: lanes 16-31 hold K 16-31

  // Preload all B fragments (W rows are [n][k] row-major == B^T, contiguous)
  ABFrag bf0[4], bf1[4];
#pragma unroll
  for (int ks = 0; ks < 4; ++ks) {
    const __bf16* p0 = w0b + (size_t)ncol * D_HID + ks * 32 + khB;
    const __bf16* p1 = w1b + (size_t)ncol * D_HID + ks * 32 + khB;
    bf0[ks].u[0] = reinterpret_cast<const uint4*>(p0)[0];
    bf0[ks].u[1] = reinterpret_cast<const uint4*>(p0)[1];
    bf1[ks].u[0] = reinterpret_cast<const uint4*>(p1)[0];
    bf1[ks].u[1] = reinterpret_cast<const uint4*>(p1)[1];
  }

  v8f acc0 = {}; v8f acc1 = {};
#pragma unroll
  for (int ks = 0; ks < 4; ++ks) {
    // A fragment, contiguous K order (same as B frag):
    //   elems 0..7  <- K = kb + khA + 0..7
    //   elems 8..15 <- K = kb + khA + 8..15
    ABFrag a;
    const __bf16* pa = &atile[m * 136 + ks * 32 + khA];
    a.u[0] = *reinterpret_cast<const uint4*>(pa);
    a.u[1] = *reinterpret_cast<const uint4*>(pa + 8);
    acc0 = __builtin_amdgcn_wmma_f32_16x16x32_bf16(
        false, a.v, false, bf0[ks].v, (short)0, acc0, false, false);
    acc1 = __builtin_amdgcn_wmma_f32_16x16x32_bf16(
        false, a.v, false, bf1[ks].v, (short)0, acc1, false, false);
  }

  // C/D layout: lane half selects M base (0 or 8); VGPR i -> row base+i.
  // Hoisted tile-in-range check: with N % 16 == 0 every block takes the
  // straight-line path (no per-row exec-mask churn).
  const float bias0 = b0[ncol];
  const float bias1 = b1[ncol];
  const int   rbase = row0 + (hi << 3);
  float* __restrict__ yp = ybuf + (size_t)rbase * D_HID + ncol;
  float* __restrict__ np = nbuf + (size_t)rbase * D_HID + ncol;
  if (row0 + 16 <= nrows) {
#pragma unroll
    for (int i = 0; i < 8; ++i) {
      yp[(size_t)i * D_HID] = acc0[i] + bias0;
      np[(size_t)i * D_HID] = acc1[i] + bias1;
    }
  } else {
#pragma unroll
    for (int i = 0; i < 8; ++i) {
      if (rbase + i < nrows) {
        yp[(size_t)i * D_HID] = acc0[i] + bias0;
        np[(size_t)i * D_HID] = acc1[i] + bias1;
      }
    }
  }
}

// ---------------------------------------------------------------------------
// Edge scatter: out[i] += n[j]; out[j] += n[i].  32 threads / edge (4 f32 ea).
// ---------------------------------------------------------------------------
__global__ __launch_bounds__(256) void k_scatter(const int* __restrict__ edges,
                                                 const float* __restrict__ nbuf,
                                                 float* __restrict__ out, int E)
{
  const int t = blockIdx.x * blockDim.x + threadIdx.x;
  const int e = t >> 5;
  if (e >= E) return;
  const int q  = t & 31;
  const int2 ij = reinterpret_cast<const int2*>(edges)[e];
  const float4 nj = reinterpret_cast<const float4*>(nbuf + (size_t)ij.y * D_HID)[q];
  const float4 ni = reinterpret_cast<const float4*>(nbuf + (size_t)ij.x * D_HID)[q];
  float* oi = out + (size_t)ij.x * D_HID + q * 4;
  float* oj = out + (size_t)ij.y * D_HID + q * 4;
  __hip_atomic_fetch_add(oi + 0, nj.x, __ATOMIC_RELAXED, __HIP_MEMORY_SCOPE_AGENT);
  __hip_atomic_fetch_add(oi + 1, nj.y, __ATOMIC_RELAXED, __HIP_MEMORY_SCOPE_AGENT);
  __hip_atomic_fetch_add(oi + 2, nj.z, __ATOMIC_RELAXED, __HIP_MEMORY_SCOPE_AGENT);
  __hip_atomic_fetch_add(oi + 3, nj.w, __ATOMIC_RELAXED, __HIP_MEMORY_SCOPE_AGENT);
  __hip_atomic_fetch_add(oj + 0, ni.x, __ATOMIC_RELAXED, __HIP_MEMORY_SCOPE_AGENT);
  __hip_atomic_fetch_add(oj + 1, ni.y, __ATOMIC_RELAXED, __HIP_MEMORY_SCOPE_AGENT);
  __hip_atomic_fetch_add(oj + 2, ni.z, __ATOMIC_RELAXED, __HIP_MEMORY_SCOPE_AGENT);
  __hip_atomic_fetch_add(oj + 3, ni.w, __ATOMIC_RELAXED, __HIP_MEMORY_SCOPE_AGENT);
}

// ---------------------------------------------------------------------------
// relu in place + bf16 copy for next layer; optional extra f32 tee
// ---------------------------------------------------------------------------
__global__ __launch_bounds__(256) void k_relu_cvt(float* __restrict__ buf,
                                                  __bf16* __restrict__ xb,
                                                  float* __restrict__ extra,
                                                  int count4)
{
  const int t = blockIdx.x * blockDim.x + threadIdx.x;
  if (t >= count4) return;
  float4 v = reinterpret_cast<float4*>(buf)[t];
  v.x = fmaxf(v.x, 0.f); v.y = fmaxf(v.y, 0.f);
  v.z = fmaxf(v.z, 0.f); v.w = fmaxf(v.w, 0.f);
  reinterpret_cast<float4*>(buf)[t] = v;
  BF4 o;
  o.h[0] = (__bf16)v.x; o.h[1] = (__bf16)v.y;
  o.h[2] = (__bf16)v.z; o.h[3] = (__bf16)v.w;
  reinterpret_cast<uint2*>(xb)[t] = o.u;
  if (extra) reinterpret_cast<float4*>(extra)[t] = v;
}

// s = residual + x, emitted as bf16 input for the final layer
__global__ __launch_bounds__(256) void k_addres_cvt(const float* __restrict__ res,
                                                    const float* __restrict__ x,
                                                    __bf16* __restrict__ xb,
                                                    int count4)
{
  const int t = blockIdx.x * blockDim.x + threadIdx.x;
  if (t >= count4) return;
  const float4 a = reinterpret_cast<const float4*>(res)[t];
  const float4 b = reinterpret_cast<const float4*>(x)[t];
  BF4 o;
  o.h[0] = (__bf16)(a.x + b.x); o.h[1] = (__bf16)(a.y + b.y);
  o.h[2] = (__bf16)(a.z + b.z); o.h[3] = (__bf16)(a.w + b.w);
  reinterpret_cast<uint2*>(xb)[t] = o.u;
}

// ---------------------------------------------------------------------------
// Final 128 -> 3 dual transform (0.2 GFLOP total: scalar is fine)
// ---------------------------------------------------------------------------
__global__ __launch_bounds__(256) void k_gemm3(const __bf16* __restrict__ xb,
                                               const float* __restrict__ w0,
                                               const float* __restrict__ b0,
                                               const float* __restrict__ w1,
                                               const float* __restrict__ b1,
                                               float* __restrict__ y3,
                                               float* __restrict__ n3, int N)
{
  const int m = blockIdx.x * blockDim.x + threadIdx.x;
  if (m >= N) return;
  float a0 = b0[0], a1 = b0[1], a2 = b0[2];
  float c0 = b1[0], c1 = b1[1], c2 = b1[2];
  const __bf16* xr = xb + (size_t)m * D_HID;
#pragma unroll 4
  for (int k = 0; k < D_HID; ++k) {
    const float xv = (float)xr[k];
    a0 = fmaf(xv, w0[k],             a0);
    a1 = fmaf(xv, w0[D_HID + k],     a1);
    a2 = fmaf(xv, w0[2 * D_HID + k], a2);
    c0 = fmaf(xv, w1[k],             c0);
    c1 = fmaf(xv, w1[D_HID + k],     c1);
    c2 = fmaf(xv, w1[2 * D_HID + k], c2);
  }
  y3[3 * (size_t)m + 0] = a0; y3[3 * (size_t)m + 1] = a1; y3[3 * (size_t)m + 2] = a2;
  n3[3 * (size_t)m + 0] = c0; n3[3 * (size_t)m + 1] = c1; n3[3 * (size_t)m + 2] = c2;
}

__global__ __launch_bounds__(256) void k_scatter3(const int* __restrict__ edges,
                                                  const float* __restrict__ n3,
                                                  float* __restrict__ out, int E)
{
  const int e = blockIdx.x * blockDim.x + threadIdx.x;
  if (e >= E) return;
  const int2 ij = reinterpret_cast<const int2*>(edges)[e];
#pragma unroll
  for (int r = 0; r < 3; ++r) {
    __hip_atomic_fetch_add(out + 3 * (size_t)ij.x + r, n3[3 * (size_t)ij.y + r],
                           __ATOMIC_RELAXED, __HIP_MEMORY_SCOPE_AGENT);
    __hip_atomic_fetch_add(out + 3 * (size_t)ij.y + r, n3[3 * (size_t)ij.x + r],
                           __ATOMIC_RELAXED, __HIP_MEMORY_SCOPE_AGENT);
  }
}

// ---------------------------------------------------------------------------
extern "C" void kernel_launch(void* const* d_in, const int* in_sizes, int n_in,
                              void* d_out, int out_size, void* d_ws, size_t ws_size,
                              hipStream_t stream) {
  (void)n_in; (void)out_size; (void)ws_size;
  const float* features = (const float*)d_in[0];
  const int*   edges    = (const int*)  d_in[1];
  const float* W0_1 = (const float*)d_in[2];  const float* b0_1 = (const float*)d_in[3];
  const float* W1_1 = (const float*)d_in[4];  const float* b1_1 = (const float*)d_in[5];
  const float* W0_h = (const float*)d_in[6];  const float* b0_h = (const float*)d_in[7];
  const float* W1_h = (const float*)d_in[8];  const float* b1_h = (const float*)d_in[9];
  const float* W0_l = (const float*)d_in[10]; const float* b0_l = (const float*)d_in[11];
  const float* W1_l = (const float*)d_in[12]; const float* b1_l = (const float*)d_in[13];

  const int N = in_sizes[0] / D_HID;            // 100000
  const int E = in_sizes[1] / 2;                // 300000
  const int L = in_sizes[6] / (D_HID * D_HID);  // 12

  // Workspace carve-out (all regions fully written before being read)
  char* ws = (char*)d_ws;
  size_t off = 0;
  auto carve = [&](size_t bytes) -> void* {
    void* p = ws + off;
    off = (off + bytes + 255) & ~(size_t)255;
    return p;
  };
  __bf16* xb   = (__bf16*)carve((size_t)N * D_HID * sizeof(__bf16));
  float*  ybuf = (float*) carve((size_t)N * D_HID * sizeof(float));
  float*  nbuf = (float*) carve((size_t)N * D_HID * sizeof(float));
  float*  res  = (float*) carve((size_t)N * D_HID * sizeof(float));
  float*  n3   = (float*) carve((size_t)N * 3 * sizeof(float));
  __bf16* wb01 = (__bf16*)carve((size_t)D_HID * D_HID * sizeof(__bf16));
  __bf16* wb11 = (__bf16*)carve((size_t)D_HID * D_HID * sizeof(__bf16));
  __bf16* wb0h = (__bf16*)carve((size_t)L * D_HID * D_HID * sizeof(__bf16));
  __bf16* wb1h = (__bf16*)carve((size_t)L * D_HID * D_HID * sizeof(__bf16));

  auto cvt = [&](const float* s, __bf16* d, int n4) {
    k_cvt_bf16<<<(n4 + 255) / 256, 256, 0, stream>>>(s, d, n4);
  };
  cvt(W0_1, wb01, D_HID * D_HID / 4);
  cvt(W1_1, wb11, D_HID * D_HID / 4);
  cvt(W0_h, wb0h, L * D_HID * D_HID / 4);
  cvt(W1_h, wb1h, L * D_HID * D_HID / 4);
  cvt(features, xb, N * D_HID / 4);

  float* vert = (float*)d_out;                 // [N,3]
  float* aux  = (float*)d_out + (size_t)N * 3; // [N,128]

  const int gGemm = (N + 15) / 16;
  const int gScat = (E * 32 + 255) / 256;
  const int c4    = N * D_HID / 4;
  const int gElem = (c4 + 255) / 256;

  // ---- layer 1 (in -> hidden), tee relu output into residual -------------
  k_gemm_dual<<<gGemm, 256, 0, stream>>>(xb, wb01, wb11, b0_1, b1_1, ybuf, nbuf, N);
  k_scatter<<<gScat, 256, 0, stream>>>(edges, nbuf, ybuf, E);
  k_relu_cvt<<<gElem, 256, 0, stream>>>(ybuf, xb, res, c4);

  // ---- 12 hidden layers; last one tees into auxiliary output -------------
  for (int l = 0; l < L; ++l) {
    k_gemm_dual<<<gGemm, 256, 0, stream>>>(
        xb, wb0h + (size_t)l * D_HID * D_HID, wb1h + (size_t)l * D_HID * D_HID,
        b0_h + (size_t)l * D_HID, b1_h + (size_t)l * D_HID, ybuf, nbuf, N);
    k_scatter<<<gScat, 256, 0, stream>>>(edges, nbuf, ybuf, E);
    k_relu_cvt<<<gElem, 256, 0, stream>>>(ybuf, xb,
                                          (l == L - 1) ? aux : (float*)nullptr, c4);
  }

  // ---- final layer on (residual + x), vertices written in-place ----------
  k_addres_cvt<<<gElem, 256, 0, stream>>>(res, ybuf, xb, c4);
  k_gemm3<<<(N + 255) / 256, 256, 0, stream>>>(xb, W0_l, b0_l, W1_l, b1_l, vert, n3, N);
  k_scatter3<<<(E + 255) / 256, 256, 0, stream>>>(edges, n3, vert, E);
}
